// MultiHeadAttentionRoPE_70145405878870
// MI455X (gfx1250) — matrix operand order repaired and checked by reading, not compiled, not run on hardware
//
#include <hip/hip_runtime.h>
#include <math.h>

#define BB 2
#define SS 2048
#define DMODEL 1024
#define NH 16
#define DH 64

typedef __attribute__((ext_vector_type(16))) __bf16 v16bf;
typedef __attribute__((ext_vector_type(8)))  float  v8f;
typedef unsigned short u16t;

// ---- helpers -------------------------------------------------------------

__device__ __forceinline__ u16t f2bf(float f) {
  union { float f; unsigned u; } x; x.f = f;
  unsigned r = x.u + 0x7FFFu + ((x.u >> 16) & 1u);   // round-to-nearest-even
  return (u16t)(r >> 16);
}

// 32 contiguous bytes (16 bf16) -> B-fragment lane data
__device__ __forceinline__ v16bf ldfrag32(const u16t* p) {
  union { v16bf v; uint4 q[2]; } u;
  u.q[0] = ((const uint4*)p)[0];
  u.q[1] = ((const uint4*)p)[1];
  return u.v;
}
// A-fragment lane data: 16 bf16 at p[0..15]
__device__ __forceinline__ v16bf ldfrag_split(const u16t* p) {
  union { v16bf v; uint4 q[2]; } u;
  u.q[0] = *(const uint4*)(p);
  u.q[1] = *(const uint4*)(p + 8);
  return u.v;
}

__device__ __forceinline__ v8f wmma_bf16(v16bf a, v16bf b, v8f c) {
  return __builtin_amdgcn_wmma_f32_16x16x32_bf16(false, a, false, b,
                                                 (short)0, c, false, false);
}

// async DMA: 16 bytes per lane, global -> LDS, tracked by ASYNCcnt
__device__ __forceinline__ void async_copy16(const u16t* g, u16t* l) {
  unsigned lds_off = (unsigned)(size_t)(l);  // aperture decode keeps [31:0]
  asm volatile("global_load_async_to_lds_b128 %0, %1, off"
               :: "v"(lds_off), "v"(g) : "memory");
}
__device__ __forceinline__ void wait_async0() {
  asm volatile("s_wait_asynccnt 0x0" ::: "memory");
}

// ---- fp32 -> bf16 convert ------------------------------------------------

__global__ void cvt_bf16_kernel(const float* __restrict__ in,
                                u16t* __restrict__ out, int n) {
  int i = blockIdx.x * blockDim.x + threadIdx.x;
  if (i < n) out[i] = f2bf(in[i]);
}

// ---- fp32 W[K,N] -> bf16 W^T[N,K] (K=N=1024) -----------------------------

__global__ void wtrans_bf16_kernel(const float* __restrict__ in,
                                   u16t* __restrict__ out) {
  int idx = blockIdx.x * blockDim.x + threadIdx.x;  // 1024*1024 threads
  int k = idx >> 10, n = idx & 1023;
  out[(size_t)n * DMODEL + k] = f2bf(in[idx]);
}

// ---- tiled WMMA GEMM: C[M,N] = A[M,K](bf16) @ Bt[N,K]^T(bf16) + bias -----
// BM=BN=128, BK=32; 8 waves/block, each wave does a 32x64 region (2x4 WMMA).
// Double-buffered LDS staging via global_load_async_to_lds_b128 (ASYNCcnt).

__global__ __launch_bounds__(256) void gemm_bf16_kernel(
    const u16t* __restrict__ A, const u16t* __restrict__ Bt,
    const float* __restrict__ bias, float* __restrict__ C,
    int M, int N, int K) {
  __shared__ __align__(16) u16t As[2][128 * 32];
  __shared__ __align__(16) u16t Bs[2][128 * 32];   // Bs[n][k]
  const int tid  = threadIdx.x;
  const int lane = tid & 31;
  const int wave = tid >> 5;
  const int bm = blockIdx.y * 128;
  const int bn = blockIdx.x * 128;
  const int wm = (wave >> 1) * 32;
  const int wn = (wave & 1) * 64;
  const int row  = tid >> 1;          // 0..127
  const int colh = (tid & 1) * 16;    // 0 or 16 (elements)

  v8f acc[2][4];
  for (int i = 0; i < 2; ++i)
    for (int j = 0; j < 4; ++j)
      for (int r = 0; r < 8; ++r) acc[i][j][r] = 0.0f;

  const int nk = K / 32;

  // prologue: stage tile 0 into buffer 0
  {
    const u16t* ga = A  + (size_t)(bm + row) * K + colh;
    const u16t* gb = Bt + (size_t)(bn + row) * K + colh;
    async_copy16(ga,     &As[0][row * 32 + colh]);
    async_copy16(ga + 8, &As[0][row * 32 + colh + 8]);
    async_copy16(gb,     &Bs[0][row * 32 + colh]);
    async_copy16(gb + 8, &Bs[0][row * 32 + colh + 8]);
  }
  wait_async0();
  __syncthreads();

  for (int kt = 0; kt < nk; ++kt) {
    const int buf = kt & 1;
    if (kt + 1 < nk) {  // prefetch next k-tile into the other buffer
      const int k0 = (kt + 1) * 32;
      const u16t* ga = A  + (size_t)(bm + row) * K + k0 + colh;
      const u16t* gb = Bt + (size_t)(bn + row) * K + k0 + colh;
      async_copy16(ga,     &As[buf ^ 1][row * 32 + colh]);
      async_copy16(ga + 8, &As[buf ^ 1][row * 32 + colh + 8]);
      async_copy16(gb,     &Bs[buf ^ 1][row * 32 + colh]);
      async_copy16(gb + 8, &Bs[buf ^ 1][row * 32 + colh + 8]);
    }

    v16bf af[2], bfr[4];
#pragma unroll
    for (int mi = 0; mi < 2; ++mi)
      af[mi] = ldfrag_split(&As[buf][(wm + 16 * mi + (lane & 15)) * 32 + 16 * (lane >> 4)]);
#pragma unroll
    for (int ni = 0; ni < 4; ++ni)
      bfr[ni] = ldfrag32(&Bs[buf][(wn + 16 * ni + (lane & 15)) * 32 + 16 * (lane >> 4)]);
#pragma unroll
    for (int mi = 0; mi < 2; ++mi)
#pragma unroll
      for (int ni = 0; ni < 4; ++ni)
        acc[mi][ni] = wmma_bf16(af[mi], bfr[ni], acc[mi][ni]);

    wait_async0();
    __syncthreads();
  }

  const int half = lane >> 4, col = lane & 15;
  for (int mi = 0; mi < 2; ++mi)
    for (int ni = 0; ni < 4; ++ni) {
      int ccol = bn + wn + 16 * ni + col;
      float bv = bias[ccol];
#pragma unroll
      for (int r = 0; r < 8; ++r) {
        int crow = bm + wm + 16 * mi + 8 * half + r;
        C[(size_t)crow * N + ccol] = acc[mi][ni][r] + bv;
      }
    }
}

// ---- RoPE: fp32 [B,S,D] -> bf16 [B,H,S,Dh], reference pairing ------------
// pair p in 0..31: angle = s * 10000^(-((2p) mod 32)/32)

__global__ void rope_kernel(const float* __restrict__ in,
                            u16t* __restrict__ out) {
  int idx = blockIdx.x * blockDim.x + threadIdx.x;  // B*S*H*32 threads
  int p = idx & 31;
  int t = idx >> 5;
  int h = t & 15;  t >>= 4;
  int s = t & 2047;
  int b = t >> 11;
  int m = (2 * p) & 31;
  float ang = (float)s * __expf(-(float)m * (9.210340371976184f / 32.0f));
  float c, sn;
  __sincosf(ang, &sn, &c);
  const float* src = in + (size_t)(b * SS + s) * DMODEL + h * DH + 2 * p;
  float x1 = src[0], x2 = src[1];
  u16t* dst = out + (size_t)((b * NH + h) * SS + s) * DH + 2 * p;
  dst[0] = f2bf(x1 * c - x2 * sn);
  dst[1] = f2bf(x1 * sn + x2 * c);
}

// ---- V: fp32 [B,S,D] -> bf16 [B,H,Dh,S] (dh-major for P@V B-fragments) ---

__global__ void vtrans_kernel(const float* __restrict__ in,
                              u16t* __restrict__ out) {
  int idx = blockIdx.x * blockDim.x + threadIdx.x;  // B*S*D threads
  int dh = idx & 63;
  int t = idx >> 6;
  int h = t & 15;  t >>= 4;
  int s = t & 2047;
  int b = t >> 11;
  out[(size_t)((b * NH + h) * DH + dh) * SS + s] = f2bf(in[idx]);
}

// ---- flash attention: one wave per 16-query tile, 32-key steps -----------

__global__ __launch_bounds__(32) void flash_attn_kernel(
    const u16t* __restrict__ Q, const u16t* __restrict__ K,
    const u16t* __restrict__ Vt, u16t* __restrict__ O) {
  __shared__ __align__(16) u16t Pt[16 * 32];
  const int lane = threadIdx.x;
  const int q0 = blockIdx.x * 16;
  const int h = blockIdx.y, b = blockIdx.z;
  const int bh = b * NH + h;
  const int half = lane >> 4, col = lane & 15;

  v16bf qf[2];
#pragma unroll
  for (int c = 0; c < 2; ++c)
    qf[c] = ldfrag_split(Q + (size_t)(bh * SS + q0 + col) * DH + 32 * c + 16 * half);

  v8f o[4];
  for (int f = 0; f < 4; ++f)
    for (int r = 0; r < 8; ++r) o[f][r] = 0.0f;
  float mrow[8], lrow[8];
  for (int r = 0; r < 8; ++r) { mrow[r] = -1e30f; lrow[r] = 0.0f; }

  const float scale = 0.125f;  // 1/sqrt(64)

  for (int kt = 0; kt < SS; kt += 32) {
    v8f s0, s1;
    for (int r = 0; r < 8; ++r) { s0[r] = 0.0f; s1[r] = 0.0f; }
#pragma unroll
    for (int c = 0; c < 2; ++c) {
      v16bf kA = ldfrag32(K + (size_t)(bh * SS + kt + col) * DH + 32 * c + 16 * half);
      v16bf kB = ldfrag32(K + (size_t)(bh * SS + kt + 16 + col) * DH + 32 * c + 16 * half);
      s0 = wmma_bf16(qf[c], kA, s0);
      s1 = wmma_bf16(qf[c], kB, s1);
    }
    // online softmax; fragment row = 8*half + r, col = lane&15 (key)
#pragma unroll
    for (int r = 0; r < 8; ++r) {
      float a0 = s0[r] * scale, a1 = s1[r] * scale;
      float mx = fmaxf(a0, a1);
#pragma unroll
      for (int off = 1; off < 16; off <<= 1)
        mx = fmaxf(mx, __shfl_xor(mx, off, 32));
      float mnew = fmaxf(mrow[r], mx);
      float corr = __expf(mrow[r] - mnew);
      float p0 = __expf(a0 - mnew), p1 = __expf(a1 - mnew);
      float rs = p0 + p1;
#pragma unroll
      for (int off = 1; off < 16; off <<= 1)
        rs += __shfl_xor(rs, off, 32);
      lrow[r] = lrow[r] * corr + rs;
      mrow[r] = mnew;
#pragma unroll
      for (int f = 0; f < 4; ++f) o[f][r] *= corr;
      s0[r] = p0; s1[r] = p1;
    }
    // transpose P (C-layout) -> A-layout via LDS, as bf16 16x32 row-major
#pragma unroll
    for (int r = 0; r < 8; ++r) {
      Pt[(8 * half + r) * 32 + col] = f2bf(s0[r]);
      Pt[(8 * half + r) * 32 + 16 + col] = f2bf(s1[r]);
    }
    __syncthreads();
    v16bf pf = ldfrag_split(Pt + col * 32 + 16 * half);
#pragma unroll
    for (int f = 0; f < 4; ++f) {
      v16bf vf = ldfrag32(Vt + (size_t)(bh * DH + 16 * f + col) * SS + kt + 16 * half);
      o[f] = wmma_bf16(pf, vf, o[f]);
    }
    __syncthreads();
  }

#pragma unroll
  for (int r = 0; r < 8; ++r) {
    float inv = 1.0f / lrow[r];
    int row = q0 + 8 * half + r;
    u16t* dst = O + (size_t)(b * SS + row) * DMODEL + h * DH;
#pragma unroll
    for (int f = 0; f < 4; ++f)
      dst[16 * f + col] = f2bf(o[f][r] * inv);
  }
}

// ---- launch --------------------------------------------------------------

extern "C" void kernel_launch(void* const* d_in, const int* in_sizes, int n_in,
                              void* d_out, int out_size, void* d_ws, size_t ws_size,
                              hipStream_t stream) {
  const float* x  = (const float*)d_in[0];
  const float* Wq = (const float*)d_in[1];
  const float* bq = (const float*)d_in[2];
  const float* Wk = (const float*)d_in[3];
  const float* bk = (const float*)d_in[4];
  const float* Wv = (const float*)d_in[5];
  const float* bv = (const float*)d_in[6];
  const float* Wo = (const float*)d_in[7];
  const float* bo = (const float*)d_in[8];
  float* out = (float*)d_out;

  char* ws = (char*)d_ws;
  const size_t MB = 1u << 20;
  u16t*  xbf  = (u16t*)(ws + 0);        // 8 MB
  u16t*  wqb  = (u16t*)(ws + 8 * MB);   // 2 MB each (transposed bf16)
  u16t*  wkb  = (u16t*)(ws + 10 * MB);
  u16t*  wvb  = (u16t*)(ws + 12 * MB);
  u16t*  wob  = (u16t*)(ws + 14 * MB);
  float* qf   = (float*)(ws + 16 * MB); // 16 MB each
  float* kf   = (float*)(ws + 32 * MB);
  float* vf   = (float*)(ws + 48 * MB);
  u16t*  qbf  = (u16t*)(ws + 64 * MB);  // 8 MB each
  u16t*  kbf  = (u16t*)(ws + 72 * MB);
  u16t*  vtb  = (u16t*)(ws + 80 * MB);
  u16t*  obuf = (u16t*)(ws + 16 * MB);  // reuses qf region after RoPE

  const int NTOK = BB * SS;            // 4096
  const int NX = NTOK * DMODEL;        // 4,194,304
  const int NW = DMODEL * DMODEL;      // 1,048,576

  // fp32 -> bf16 conversions (weights transposed for async-friendly tiles)
  cvt_bf16_kernel<<<NX / 256, 256, 0, stream>>>(x, xbf, NX);
  wtrans_bf16_kernel<<<NW / 256, 256, 0, stream>>>(Wq, wqb);
  wtrans_bf16_kernel<<<NW / 256, 256, 0, stream>>>(Wk, wkb);
  wtrans_bf16_kernel<<<NW / 256, 256, 0, stream>>>(Wv, wvb);
  wtrans_bf16_kernel<<<NW / 256, 256, 0, stream>>>(Wo, wob);

  // QKV projections (WMMA GEMM, async double-buffered staging)
  dim3 ggrid(DMODEL / 128, NTOK / 128);
  gemm_bf16_kernel<<<ggrid, 256, 0, stream>>>(xbf, wqb, bq, qf, NTOK, DMODEL, DMODEL);
  gemm_bf16_kernel<<<ggrid, 256, 0, stream>>>(xbf, wkb, bk, kf, NTOK, DMODEL, DMODEL);
  gemm_bf16_kernel<<<ggrid, 256, 0, stream>>>(xbf, wvb, bv, vf, NTOK, DMODEL, DMODEL);

  // RoPE + layout transforms
  int nrope = BB * SS * NH * 32;
  rope_kernel<<<nrope / 256, 256, 0, stream>>>(qf, qbf);
  rope_kernel<<<nrope / 256, 256, 0, stream>>>(kf, kbf);
  vtrans_kernel<<<NX / 256, 256, 0, stream>>>(vf, vtb);

  // flash attention (WMMA for QK^T and P@V)
  dim3 agrid(SS / 16, NH, BB);
  flash_attn_kernel<<<agrid, 32, 0, stream>>>(qbf, kbf, vtb, obuf);

  // output projection -> fp32 d_out
  gemm_bf16_kernel<<<ggrid, 256, 0, stream>>>(obuf, wob, bo, out, NTOK, DMODEL, DMODEL);
}
